// Attention_87282325389655
// MI455X (gfx1250) — matrix operand order repaired and checked by reading, not compiled, not run on hardware
//
#include <hip/hip_runtime.h>
#include <math.h>

// ---- problem constants (from reference) ----
#define NUM_ENTITY 100000
#define BB 8192
#define NN 128
#define DD 64
#define MASK_BIG 1e19f

typedef __attribute__((ext_vector_type(16))) _Float16 v16h;
typedef __attribute__((ext_vector_type(8)))  float    v8f;

union Frag {
    v16h v;
    unsigned int u[8];
};

__global__ __launch_bounds__(256) void attn_fused_kernel(
    const float* __restrict__ input,       // (B, N, 64)
    const int*   __restrict__ neighbor,    // (B, N, 2)  [rel, ent]
    const int*   __restrict__ qrel_id,     // (B,)
    const float* __restrict__ weight,      // (B, N, 2)
    const float* __restrict__ mlp_w,       // (200001, 64)
    const float* __restrict__ qrel_w,      // (1000, 64)
    const float* __restrict__ att_w,       // (128, 128)
    const float* __restrict__ att_v,       // (1, 128)
    float* __restrict__ out_vec,           // (B, 64)
    float* __restrict__ out_att)           // (B, N)
{
    // LDS: ~69 KB -> 4 workgroups per WGP (320 KB budget)
    __shared__ __align__(16) float     trans[NN * DD];   // 32 KB f32 transformed
    __shared__ __align__(4)  _Float16  tbuf[NN * DD];    // 16 KB f16 transformed (WMMA A)
    __shared__ __align__(4)  _Float16  w2t[NN * DD];     // 16 KB f16 att_w[64:,:] transposed (WMMA B)
    __shared__ float qv[DD];
    __shared__ float qpart[NN];
    __shared__ float attv[NN];
    __shared__ float logits[NN];
    __shared__ float red[NN];
    __shared__ float aw[NN];
    __shared__ float rat[NN];
    __shared__ float po[256];
    __shared__ int   entl[NN];

    const int b   = blockIdx.x;
    const int tid = threadIdx.x;

    // ---- load query embedding ----
    if (tid < DD) {
        qv[tid] = qrel_w[(long)qrel_id[b] * DD + tid];
    }

    // ---- stage W2 = att_w[64:128, :] transposed into LDS as f16 ----
    // w2t[c*64 + k] = att_w[(64+k)*128 + c]  -> B-fragment (k,k+1) pairs contiguous
    for (int idx = tid; idx < NN * DD; idx += 256) {
        int k = idx & 63;
        int c = idx >> 6;
        w2t[c * DD + k] = (_Float16)att_w[(64 + k) * NN + c];
    }

    // ---- phase 1: per-row gather + orthogonal projection + mask ----
    // Two threads (same wave, lanes 2n / 2n+1) cooperate on each row: each
    // owns 32 of the 64 dims -> full 256-thread memory parallelism on the
    // dominant HBM stream, partial dots combined with a wave32 shfl_xor.
    {
        const int row = tid >> 1;      // 0..127
        const int hf  = tid & 1;       // which 32-float half of the row
        const long nb = ((long)b * NN + row) * 2;
        const int rel = neighbor[nb + 0];
        const int ent = neighbor[nb + 1];

        const float4* xi4 = (const float4*)(input + ((long)b * NN + row) * DD) + hf * 8;
        const float4* rr4 = (const float4*)(mlp_w + (long)rel * DD) + hf * 8;

        float4 x[8], r[8];
        float rr = 0.0f, xr = 0.0f;
        #pragma unroll
        for (int i = 0; i < 8; ++i) {
            x[i] = xi4[i];
            r[i] = rr4[i];
            rr += r[i].x * r[i].x + r[i].y * r[i].y + r[i].z * r[i].z + r[i].w * r[i].w;
            xr += x[i].x * r[i].x + x[i].y * r[i].y + x[i].z * r[i].z + x[i].w * r[i].w;
        }
        // combine the two half-row partials (partner lane = lane ^ 1)
        rr += __shfl_xor(rr, 1, 32);
        xr += __shfl_xor(xr, 1, 32);

        const float nrm  = fmaxf(sqrtf(rr), 1e-12f);
        const float coef = (xr / nrm) / nrm;          // (x . normed) / ||r||
        const float msk  = (ent < NUM_ENTITY) ? 1.0f : 0.0f;

        float4* tr4 = (float4*)(trans + row * DD) + hf * 8;
        _Float16* tb = tbuf + row * DD + hf * 32;
        #pragma unroll
        for (int i = 0; i < 8; ++i) {
            float4 t;
            t.x = (x[i].x - coef * r[i].x) * msk;
            t.y = (x[i].y - coef * r[i].y) * msk;
            t.z = (x[i].z - coef * r[i].z) * msk;
            t.w = (x[i].w - coef * r[i].w) * msk;
            tr4[i] = t;
            tb[4 * i + 0] = (_Float16)t.x;
            tb[4 * i + 1] = (_Float16)t.y;
            tb[4 * i + 2] = (_Float16)t.z;
            tb[4 * i + 3] = (_Float16)t.w;
        }

        if (hf == 0) {
            entl[row]   = ent;
            logits[row] = 0.0f;
            const float w0 = weight[nb + 0];
            const float w1 = weight[nb + 1];
            rat[row] = w0 / (w1 + 1.0f);
        }
    }
    __syncthreads();

    // ---- qpart[c] = sum_k q[k] * att_w[k, c]  (per-block constant row) ----
    if (tid < NN) {
        float acc = 0.0f;
        #pragma unroll 8
        for (int k = 0; k < DD; ++k) acc += qv[k] * att_w[k * NN + tid];
        qpart[tid] = acc;
        attv[tid]  = att_v[tid];
    }
    __syncthreads();

    // ---- phase 2: WMMA  trans(128x64) @ W2(64x128), f16 inputs, f32 acc ----
    // wave w owns rows [16w, 16w+16); accumulates all 8 column tiles.
    const int lane  = tid & 31;
    const int halfi = lane >> 4;           // K half-wave split
    const int l15   = lane & 15;
    const int strip = tid >> 5;            // wave id 0..7
    const int mrow  = (strip << 4) + l15;

    const unsigned int* tb32 = (const unsigned int*)tbuf;
    const unsigned int* w232 = (const unsigned int*)w2t;

    // A fragments for K-steps 0,1 per the 16-bit 16x32 A layout
    Frag afr[2];
    #pragma unroll
    for (int s = 0; s < 2; ++s) {
        #pragma unroll
        for (int v = 0; v < 8; ++v) {
            int kb = s * 32 + ((v >> 2) << 4) + (halfi << 3) + ((v & 3) << 1);
            afr[s].u[v] = tb32[(mrow * DD + kb) >> 1];
        }
    }

    v8f acc[8];
    #pragma unroll
    for (int t = 0; t < 8; ++t)
        acc[t] = (v8f){0.f, 0.f, 0.f, 0.f, 0.f, 0.f, 0.f, 0.f};

    #pragma unroll
    for (int t = 0; t < 8; ++t) {
        const int ncol = (t << 4) + l15;
        #pragma unroll
        for (int s = 0; s < 2; ++s) {
            Frag bfr;
            #pragma unroll
            for (int v = 0; v < 8; ++v) {
                int kb = s * 32 + ((v >> 2) << 4) + (halfi << 3) + ((v & 3) << 1);
                bfr.u[v] = w232[(ncol * DD + kb) >> 1];
            }
            acc[t] = __builtin_amdgcn_wmma_f32_16x16x32_f16(
                false, afr[s].v, false, bfr.v, (short)0, acc[t], false, false);
        }
    }

    // ---- logit[row] += sum_c tanh(acc + qpart[c]) * att_v[c] ----
    // C layout: lane -> column (t*16 + lane&15), VGPR j -> row (half*8 + j)
    float pl[8];
    #pragma unroll
    for (int j = 0; j < 8; ++j) pl[j] = 0.0f;
    #pragma unroll
    for (int t = 0; t < 8; ++t) {
        const int c  = (t << 4) + l15;
        const float qp = qpart[c];
        const float av = attv[c];
        #pragma unroll
        for (int j = 0; j < 8; ++j)
            pl[j] += tanhf(acc[t][j] + qp) * av;
    }
    const int rbase = (strip << 4) + (halfi << 3);
    #pragma unroll
    for (int j = 0; j < 8; ++j)
        atomicAdd(&logits[rbase + j], pl[j]);   // ds_add_f32
    __syncthreads();

    // ---- masked softmax over N=128 ----
    if (tid < NN) {
        float l = logits[tid];
        if (entl[tid] == NUM_ENTITY) l -= MASK_BIG;
        logits[tid] = l;
        red[tid]    = l;
    }
    __syncthreads();
    for (int s2 = 64; s2 > 0; s2 >>= 1) {
        if (tid < s2) red[tid] = fmaxf(red[tid], red[tid + s2]);
        __syncthreads();
    }
    const float mx = red[0];
    __syncthreads();
    if (tid < NN) {
        float e = expf(logits[tid] - mx);
        aw[tid]  = e;
        red[tid] = e;
    }
    __syncthreads();
    for (int s2 = 64; s2 > 0; s2 >>= 1) {
        if (tid < s2) red[tid] = red[tid] + red[tid + s2];
        __syncthreads();
    }
    const float sm = red[0];
    __syncthreads();
    if (tid < NN) {
        float w = aw[tid] / sm + rat[tid];
        aw[tid] = w;
        out_att[(long)b * NN + tid] = w;
    }
    __syncthreads();

    // ---- output[b, d] = sum_n trans[n, d] * aw[n]  (all 256 threads) ----
    {
        const int d = tid & 63;
        const int q = tid >> 6;          // quarter of the n-range
        float p = 0.0f;
        #pragma unroll 8
        for (int n = q * 32; n < q * 32 + 32; ++n)
            p += trans[n * DD + d] * aw[n];
        po[tid] = p;
    }
    __syncthreads();
    if (tid < DD) {
        out_vec[(long)b * DD + tid] =
            po[tid] + po[tid + 64] + po[tid + 128] + po[tid + 192];
    }
}

extern "C" void kernel_launch(void* const* d_in, const int* in_sizes, int n_in,
                              void* d_out, int out_size, void* d_ws, size_t ws_size,
                              hipStream_t stream) {
    const float* input    = (const float*)d_in[0];
    const int*   neighbor = (const int*)  d_in[1];
    const int*   qid      = (const int*)  d_in[2];
    const float* weight   = (const float*)d_in[3];
    const float* mlp_w    = (const float*)d_in[4];
    const float* qrel_w   = (const float*)d_in[5];
    const float* att_w    = (const float*)d_in[6];
    const float* att_v    = (const float*)d_in[7];

    float* out     = (float*)d_out;
    float* out_vec = out;                       // (B, 64) first in tuple
    float* out_att = out + (size_t)BB * DD;     // (B, 128) second

    attn_fused_kernel<<<dim3(BB), dim3(256), 0, stream>>>(
        input, neighbor, qid, weight, mlp_w, qrel_w, att_w, att_v,
        out_vec, out_att);
}
